// PGBF_Surv_78228534329902
// MI455X (gfx1250) — matrix operand order repaired and checked by reading, not compiled, not run on hardware
//
#include <hip/hip_runtime.h>
#include <hip/hip_bf16.h>
#include <stdint.h>

// ---------- types ----------
typedef __attribute__((ext_vector_type(16))) __bf16          v16bf;
typedef __attribute__((ext_vector_type(8)))  float           v8f;
typedef __attribute__((ext_vector_type(16))) unsigned short  us16;
typedef __attribute__((ext_vector_type(8)))  unsigned short  us8;
typedef __attribute__((ext_vector_type(4)))  unsigned short  us4;
typedef __attribute__((ext_vector_type(4)))  float           f4;

union ABf { us16 u; v16bf b; us8 h[2]; };

static __device__ __forceinline__ unsigned short f2bf(float f) {
  unsigned int u = __float_as_uint(f);
  u += 0x7fffu + ((u >> 16) & 1u);          // round-to-nearest-even
  return (unsigned short)(u >> 16);
}

#define NNODES 8192
#define DD     256
#define KK     30

// ---------- fp32 -> bf16 (no transpose; for activations) ----------
__global__ void cvt_kernel(const float* __restrict__ in, unsigned short* __restrict__ out, int n4) {
  int i = blockIdx.x * blockDim.x + threadIdx.x;
  if (i < n4) {
    f4 v = *(const f4*)&in[(size_t)i * 4];
    us4 o;
#pragma unroll
    for (int q = 0; q < 4; ++q) o[q] = f2bf(v[q]);
    *(us4*)&out[(size_t)i * 4] = o;
  }
}

// ---------- fp32 [K,N] -> bf16 transposed [N,K] (for weights, LDS-tiled) ----------
__global__ __launch_bounds__(256) void cvtT_kernel(const float* __restrict__ in,
                                                   unsigned short* __restrict__ out,
                                                   int K, int N) {
  __shared__ float tile[32][33];
  const int bx = blockIdx.x * 32;  // n offset
  const int by = blockIdx.y * 32;  // k offset
  const int tx = threadIdx.x & 31, ty = threadIdx.x >> 5;
#pragma unroll
  for (int i = 0; i < 4; ++i)
    tile[ty + 8 * i][tx] = in[(size_t)(by + ty + 8 * i) * N + bx + tx];
  __syncthreads();
#pragma unroll
  for (int i = 0; i < 4; ++i)
    out[(size_t)(bx + ty + 8 * i) * K + by + tx] = f2bf(tile[tx][ty + 8 * i]);
}

// ---------- bf16 WMMA GEMM: C = act(A[M,K] * Bt[N,K]^T + bias) ----------
// 256 threads = 8 waves; block tile 64x32; double-buffered LDS; dual accumulators
// (unroll-by-2) so consecutive WMMAs have no RAW dependence.
template <int ACT, bool DO_ACC, bool WR_F, bool WR_BF>
__global__ __launch_bounds__(256) void gemm_bf16_kernel(
    const unsigned short* __restrict__ A, const unsigned short* __restrict__ Bt,
    const float* __restrict__ bias, float* __restrict__ Cf,
    unsigned short* __restrict__ Cbf, int M, int N, int Kd) {
  __shared__ unsigned short As[2][64][32];
  __shared__ unsigned short Bs[2][32][32];   // Bs[buf][n][k]
  const int tid  = threadIdx.x;
  const int lane = tid & 31, wave = tid >> 5;
  const int wm = wave >> 1, wn = wave & 1;
  const int mBase = blockIdx.y * 64, nBase = blockIdx.x * 32;
  const int ar = tid >> 2, ac8 = (tid & 3) * 8;  // A: 16B per thread
  const int bn = tid >> 3, bk4 = (tid & 7) * 4;  // B: 8B per thread
  const int nk = Kd >> 5;                        // always even (32 or 8)

  auto load_tile = [&](int kt, int buf) {
    const int k0 = kt * 32;
    *(us8*)&As[buf][ar][ac8] = *(const us8*)&A[(size_t)(mBase + ar) * Kd + k0 + ac8];
    *(us4*)&Bs[buf][bn][bk4] = *(const us4*)&Bt[(size_t)(nBase + bn) * Kd + k0 + bk4];
  };
  load_tile(0, 0);

  v8f acc0 = {}, acc1 = {};
  const int row = wm * 16 + (lane & 15);
  const int hi  = lane >> 4;
  const int cn  = wn * 16 + (lane & 15);
  for (int kt = 0; kt < nk; kt += 2) {
    __syncthreads();                       // tile kt (buf0) ready
    load_tile(kt + 1, 1);                  // prefetch odd tile
    {
      ABf a, b;
      a.u = *(const us16*)&As[0][row][hi * 16];        // A 16x32 frag layout
      b.u = *(const us16*)&Bs[0][cn][hi * 16];         // B 32x16 frag layout
      acc0 = __builtin_amdgcn_wmma_f32_16x16x32_bf16(false, a.b, false, b.b,
                                                     (short)0, acc0, false, false);
    }
    __syncthreads();                       // tile kt+1 (buf1) ready, buf0 reads done
    if (kt + 2 < nk) load_tile(kt + 2, 0); // prefetch next even tile
    {
      ABf a, b;
      a.u = *(const us16*)&As[1][row][hi * 16];
      b.u = *(const us16*)&Bs[1][cn][hi * 16];
      acc1 = __builtin_amdgcn_wmma_f32_16x16x32_bf16(false, a.b, false, b.b,
                                                     (short)0, acc1, false, false);
    }
  }
  const v8f acc = acc0 + acc1;
  const float bvv = bias[nBase + cn];
  const int gn = nBase + cn;
#pragma unroll
  for (int r = 0; r < 8; ++r) {            // C layout: VGPR r -> M = r + hi*8
    const int m = mBase + wm * 16 + r + hi * 8;
    float v = acc[r] + bvv;
    if constexpr (ACT == 1) v = fmaxf(v, 0.f);
    if constexpr (ACT == 2) v = v > 0.f ? v : 0.01f * v;
    const size_t idx = (size_t)m * N + gn;
    if constexpr (DO_ACC) v += Cf[idx];
    if constexpr (WR_F)  Cf[idx] = v;
    if constexpr (WR_BF) Cbf[idx] = f2bf(v);
  }
}

// ---------- e_t row sums ----------
__global__ __launch_bounds__(256) void rowsum_kernel(const float* __restrict__ x,
                                                     float* __restrict__ out) {
  int wave = threadIdx.x >> 5, lane = threadIdx.x & 31;
  int row = blockIdx.x * 8 + wave;
  f4 a = *(const f4*)&x[(size_t)row * DD + lane * 8];
  f4 b = *(const f4*)&x[(size_t)row * DD + lane * 8 + 4];
  float s = (a[0] + a[1]) + (a[2] + a[3]) + (b[0] + b[1]) + (b[2] + b[3]);
#pragma unroll
  for (int off = 16; off > 0; off >>= 1) s += __shfl_xor(s, off, 32);
  if (!lane) out[row] = s;
}

// ---------- fused scores + top-30 + gated neighbor aggregation ----------
// block = 256 threads (8 waves) owns 16 rows; scores never hit HBM.
__global__ __launch_bounds__(256) void attn_kernel(
    const unsigned short* __restrict__ eh_bf, const float* __restrict__ eh_f,
    const unsigned short* __restrict__ et_bf, const float* __restrict__ et_f,
    const float* __restrict__ et_rs,
    unsigned short* __restrict__ s_bf, unsigned short* __restrict__ b_bf) {
  __shared__ unsigned short ehs[16][256];   // A rows (bf16), staged once
  __shared__ float scores[16][128];         // reused later as ka_weight storage
  __shared__ float tval[16][KK];
  __shared__ int   tidxs[16][KK];
  __shared__ float kap[16][KK];
  __shared__ unsigned short cidx[16][128];  // filtered candidate list
  __shared__ int   ccnt[16];
  __shared__ float rmin[16];
  __shared__ int   rcnt[16];

  const int tid  = threadIdx.x;
  const int lane = tid & 31, wave = tid >> 5;
  const int row0 = blockIdx.x * 16;

  { // stage 16 e_h rows into LDS
    int e0 = tid * 16, r = e0 >> 8, c = e0 & 255;
    *(us8*)&ehs[r][c]     = *(const us8*)&eh_bf[(size_t)(row0 + r) * DD + c];
    *(us8*)&ehs[r][c + 8] = *(const us8*)&eh_bf[(size_t)(row0 + r) * DD + c + 8];
  }
  if (tid < 16) { ccnt[tid] = 0; rmin[tid] = -3.4e38f; rcnt[tid] = 0; }
  __syncthreads();

  const int hi  = lane >> 4;
  const int l15 = lane & 15;

  // hoist chunk-invariant A fragments into registers (8 k-steps x 32B)
  ABf afr[8];
#pragma unroll
  for (int ks = 0; ks < 8; ++ks) {
    const int k0 = ks * 32;
    afr[ks].u = *(const us16*)&ehs[l15][k0 + hi * 16];
  }

  int cnt = 0, minp = 0;
  float minv = -3.4e38f;
  const int frow = tid >> 4, fsub = tid & 15;   // filter stage: 16 threads/row

  for (int chunk = 0; chunk < NNODES / 128; ++chunk) {
    const int nbase = chunk * 128 + wave * 16;
    const unsigned short* bcol = &et_bf[(size_t)(nbase + l15) * DD + hi * 16];
    v8f acc0 = {}, acc1 = {};    // dual accumulators: no WMMA->WMMA RAW chain
#pragma unroll
    for (int ks = 0; ks < 8; ks += 2) {
      ABf b0, b1;   // B = e_t^T: 16 contiguous bf16 of row j at offset k (L2-resident)
      b0.u = *(const us16*)&bcol[ks * 32];
      b1.u = *(const us16*)&bcol[ks * 32 + 32];
      acc0 = __builtin_amdgcn_wmma_f32_16x16x32_bf16(false, afr[ks].b, false, b0.b,
                                                     (short)0, acc0, false, false);
      acc1 = __builtin_amdgcn_wmma_f32_16x16x32_bf16(false, afr[ks + 1].b, false, b1.b,
                                                     (short)0, acc1, false, false);
    }
    const v8f acc = acc0 + acc1;
#pragma unroll
    for (int r = 0; r < 8; ++r)
      scores[r + hi * 8][wave * 16 + l15] = acc[r] * 0.0625f;   // SCALE = D^-0.5
    __syncthreads();

    { // stage 1: parallel threshold filter + compaction (LDS atomics)
      const float th = rmin[frow];
      const bool fill = rcnt[frow] < KK;
#pragma unroll
      for (int q = 0; q < 8; ++q) {
        const int j = fsub * 8 + q;
        const float v = scores[frow][j];
        if (fill || v > th) {
          int pos = atomicAdd(&ccnt[frow], 1);
          cidx[frow][pos] = (unsigned short)j;
        }
      }
    }
    __syncthreads();
    if (tid < 16) {  // stage 2: serial insertion of the few survivors
      const int nc = ccnt[tid];
      const int colbase = chunk * 128;
      for (int i = 0; i < nc; ++i) {
        const int j = cidx[tid][i];
        const float v = scores[tid][j];
        if (cnt < KK) {
          tval[tid][cnt] = v; tidxs[tid][cnt] = colbase + j; ++cnt;
          if (cnt == KK) {
            minv = tval[tid][0]; minp = 0;
            for (int q = 1; q < KK; ++q)
              if (tval[tid][q] < minv) { minv = tval[tid][q]; minp = q; }
          }
        } else if (v > minv) {
          tval[tid][minp] = v; tidxs[tid][minp] = colbase + j;
          minv = tval[tid][0]; minp = 0;
          for (int q = 1; q < KK; ++q)
            if (tval[tid][q] < minv) { minv = tval[tid][q]; minp = q; }
        }
      }
      ccnt[tid] = 0; rmin[tid] = minv; rcnt[tid] = cnt;
    }
    __syncthreads();
  }

  if (tid < 16) {  // softmax over the 30 kept logits -> probabilities in tval
    float mx = tval[tid][0];
    for (int q = 1; q < KK; ++q) mx = fmaxf(mx, tval[tid][q]);
    float s = 0.f;
    for (int q = 0; q < KK; ++q) { float e = __expf(tval[tid][q] - mx); tval[tid][q] = e; s += e; }
    float inv = 1.f / s;
    for (int q = 0; q < KK; ++q) tval[tid][q] *= inv;
  }
  __syncthreads();

  // gated aggregation: wave per row, lane owns 8 of 256 channels
  const int d0 = lane * 8;
  for (int rr = 0; rr < 2; ++rr) {
    const int m = rr * 8 + wave;
    const int row = row0 + m;
    float eh[8];
#pragma unroll
    for (int i = 0; i < 8; ++i) eh[i] = eh_f[(size_t)row * DD + d0 + i];
    // ka_weight[k] = rowsum(e_t[j]) * sum_d tanh((2-p)*eh + p*et)
    for (int k = 0; k < KK; ++k) {
      const float p = tval[m][k];
      const int j = tidxs[m][k];
      const float* et = &et_f[(size_t)j * DD + d0];
      float s = 0.f;
#pragma unroll
      for (int i = 0; i < 8; ++i) {
        float x = (2.f - p) * eh[i] + p * et[i];
        x = fminf(fmaxf(x, -10.f), 10.f);
        float e2 = __expf(2.f * x);
        s += (e2 - 1.f) / (e2 + 1.f);
      }
#pragma unroll
      for (int off = 16; off > 0; off >>= 1) s += __shfl_xor(s, off, 32);
      if (lane == 0) scores[m][k] = et_rs[j] * s;
    }
    __syncthreads();
    { // softmax over KK per row
      float v = (lane < KK) ? scores[m][lane] : -3.4e38f;
      float mx = v;
#pragma unroll
      for (int off = 16; off > 0; off >>= 1) mx = fmaxf(mx, __shfl_xor(mx, off, 32));
      float e = (lane < KK) ? __expf(v - mx) : 0.f;
      float s = e;
#pragma unroll
      for (int off = 16; off > 0; off >>= 1) s += __shfl_xor(s, off, 32);
      if (lane < KK) kap[m][lane] = e / s;
    }
    __syncthreads();
    float a8[8] = {};
    for (int k = 0; k < KK; ++k) {
      const float p2 = kap[m][k];
      const int j = tidxs[m][k];
      const float* et = &et_f[(size_t)j * DD + d0];
#pragma unroll
      for (int i = 0; i < 8; ++i) a8[i] += p2 * et[i];
    }
    us8 sv, bv;
#pragma unroll
    for (int i = 0; i < 8; ++i) {
      sv[i] = f2bf(eh[i] + a8[i]);
      bv[i] = f2bf(eh[i] * a8[i]);
    }
    *(us8*)&s_bf[(size_t)row * DD + d0] = sv;
    *(us8*)&b_bf[(size_t)row * DD + d0] = bv;
    __syncthreads();
  }
}

// ---------- att = z . W_att2 + b ----------
__global__ __launch_bounds__(256) void attdot_kernel(const float* __restrict__ z,
                                                     const float* __restrict__ w2,
                                                     const float* __restrict__ b2,
                                                     float* __restrict__ att) {
  int wave = threadIdx.x >> 5, lane = threadIdx.x & 31;
  int row = blockIdx.x * 8 + wave;
  f4 zv = *(const f4*)&z[(size_t)row * 128 + lane * 4];
  f4 wv = *(const f4*)&w2[lane * 4];
  float s = zv[0] * wv[0] + zv[1] * wv[1] + zv[2] * wv[2] + zv[3] * wv[3];
#pragma unroll
  for (int off = 16; off > 0; off >>= 1) s += __shfl_xor(s, off, 32);
  if (!lane) att[row] = s + b2[0];
}

// ---------- global softmax over N + weighted readout e_g ----------
__global__ __launch_bounds__(1024) void softmax_eg_kernel(const float* __restrict__ att,
                                                          const float* __restrict__ e,
                                                          float* __restrict__ eg) {
  __shared__ float a_sh[NNODES];
  __shared__ float red[1024];
  __shared__ float s_mx, s_sum;
  int t = threadIdx.x;
  float lm = -3.4e38f;
#pragma unroll
  for (int i = 0; i < 8; ++i) lm = fmaxf(lm, att[t + i * 1024]);
  red[t] = lm; __syncthreads();
  for (int s = 512; s > 0; s >>= 1) { if (t < s) red[t] = fmaxf(red[t], red[t + s]); __syncthreads(); }
  if (t == 0) s_mx = red[0];
  __syncthreads();
  float mx = s_mx, ls = 0.f;
#pragma unroll
  for (int i = 0; i < 8; ++i) {
    int n = t + i * 1024;
    float ev = __expf(att[n] - mx);
    a_sh[n] = ev; ls += ev;
  }
  red[t] = ls; __syncthreads();
  for (int s = 512; s > 0; s >>= 1) { if (t < s) red[t] += red[t + s]; __syncthreads(); }
  if (t == 0) s_sum = red[0];
  __syncthreads();
  float inv = 1.f / s_sum;
  int d = t & 255, part = t >> 8;
  float acc = 0.f;
  for (int n = part * 2048; n < (part + 1) * 2048; ++n)
    acc += a_sh[n] * e[(size_t)n * DD + d];
  red[t] = acc; __syncthreads();
  if (t < 256) eg[d] = (red[t] + red[t + 256] + red[t + 512] + red[t + 768]) * inv;
}

// ---------- host ----------
extern "C" void kernel_launch(void* const* d_in, const int* in_sizes, int n_in,
                              void* d_out, int out_size, void* d_ws, size_t ws_size,
                              hipStream_t stream) {
  (void)in_sizes; (void)n_in; (void)out_size; (void)ws_size;
  const float* x      = (const float*)d_in[0];
  const float* Wfc1   = (const float*)d_in[1];  const float* bfc1 = (const float*)d_in[2];
  const float* Wfc2   = (const float*)d_in[3];  const float* bfc2 = (const float*)d_in[4];
  const float* Whead  = (const float*)d_in[5];  const float* bhead= (const float*)d_in[6];
  const float* Wtail  = (const float*)d_in[7];  const float* btail= (const float*)d_in[8];
  const float* Wlin1  = (const float*)d_in[9];  const float* blin1= (const float*)d_in[10];
  const float* Wlin2  = (const float*)d_in[11]; const float* blin2= (const float*)d_in[12];
  const float* Watt1  = (const float*)d_in[13]; const float* batt1= (const float*)d_in[14];
  const float* Watt2  = (const float*)d_in[15]; const float* batt2= (const float*)d_in[16];

  char* p = (char*)d_ws;
  auto carve = [&](size_t bytes) { char* r = p; p += (bytes + 255) & ~size_t(255); return r; };
  unsigned short* x_bf  = (unsigned short*)carve((size_t)NNODES * 1024 * 2);
  unsigned short* w1_t  = (unsigned short*)carve(1024 * 256 * 2);   // all weights bf16, transposed [N,K]
  unsigned short* w2_t  = (unsigned short*)carve(256 * 256 * 2);
  unsigned short* wh_t  = (unsigned short*)carve(256 * 256 * 2);
  unsigned short* wt_t  = (unsigned short*)carve(256 * 256 * 2);
  unsigned short* wl1_t = (unsigned short*)carve(256 * 256 * 2);
  unsigned short* wl2_t = (unsigned short*)carve(256 * 256 * 2);
  unsigned short* wa1_t = (unsigned short*)carve(256 * 128 * 2);
  unsigned short* h1_bf = (unsigned short*)carve((size_t)NNODES * DD * 2);
  unsigned short* h2_bf = (unsigned short*)carve((size_t)NNODES * DD * 2);
  float*          eh_f  = (float*)carve((size_t)NNODES * DD * 4);
  unsigned short* eh_bf = (unsigned short*)carve((size_t)NNODES * DD * 2);
  float*          et_f  = (float*)carve((size_t)NNODES * DD * 4);
  unsigned short* et_bf = (unsigned short*)carve((size_t)NNODES * DD * 2);
  float*          et_rs = (float*)carve(NNODES * 4);
  unsigned short* s_bf  = (unsigned short*)carve((size_t)NNODES * DD * 2);
  unsigned short* b_bf  = (unsigned short*)carve((size_t)NNODES * DD * 2);
  unsigned short* e_bf  = (unsigned short*)carve((size_t)NNODES * DD * 2);
  float*          z_f   = (float*)carve((size_t)NNODES * 128 * 4);
  float*          att_f = (float*)carve(NNODES * 4);

  float* e_out  = (float*)d_out;                       // [N, D]
  float* eg_out = (float*)d_out + (size_t)NNODES * DD; // [D]

  { int n4 = NNODES * 1024 / 4;
    cvt_kernel<<<dim3((n4 + 255) / 256), dim3(256), 0, stream>>>(x, x_bf, n4); }
  auto cvtT = [&](const float* W, unsigned short* Wt, int K, int N) {
    cvtT_kernel<<<dim3(N / 32, K / 32), dim3(256), 0, stream>>>(W, Wt, K, N);
  };
  cvtT(Wfc1, w1_t, 1024, 256);
  cvtT(Wfc2, w2_t, 256, 256);
  cvtT(Whead, wh_t, 256, 256);
  cvtT(Wtail, wt_t, 256, 256);
  cvtT(Wlin1, wl1_t, 256, 256);
  cvtT(Wlin2, wl2_t, 256, 256);
  cvtT(Watt1, wa1_t, 256, 128);

  // <ACT, DO_ACC, WR_F, WR_BF>
  gemm_bf16_kernel<1, false, false, true><<<dim3(256 / 32, NNODES / 64), dim3(256), 0, stream>>>(
      x_bf, w1_t, bfc1, nullptr, h1_bf, NNODES, 256, 1024);                        // h = relu(fc1)
  gemm_bf16_kernel<1, false, false, true><<<dim3(256 / 32, NNODES / 64), dim3(256), 0, stream>>>(
      h1_bf, w2_t, bfc2, nullptr, h2_bf, NNODES, 256, 256);                        // h2 = relu(fc2)
  gemm_bf16_kernel<0, false, true, true><<<dim3(256 / 32, NNODES / 64), dim3(256), 0, stream>>>(
      h2_bf, wh_t, bhead, eh_f, eh_bf, NNODES, 256, 256);                          // e_h
  gemm_bf16_kernel<0, false, true, true><<<dim3(256 / 32, NNODES / 64), dim3(256), 0, stream>>>(
      h2_bf, wt_t, btail, et_f, et_bf, NNODES, 256, 256);                          // e_t

  rowsum_kernel<<<dim3(NNODES / 8), dim3(256), 0, stream>>>(et_f, et_rs);

  attn_kernel<<<dim3(NNODES / 16), dim3(256), 0, stream>>>(
      eh_bf, eh_f, et_bf, et_f, et_rs, s_bf, b_bf);

  gemm_bf16_kernel<1, false, true, false><<<dim3(256 / 32, NNODES / 64), dim3(256), 0, stream>>>(
      s_bf, wl1_t, blin1, e_out, nullptr, NNODES, 256, 256);                       // sum_emb
  gemm_bf16_kernel<1, true, true, false><<<dim3(256 / 32, NNODES / 64), dim3(256), 0, stream>>>(
      b_bf, wl2_t, blin2, e_out, nullptr, NNODES, 256, 256);                       // e = sum+bi

  { int n4 = NNODES * 256 / 4;
    cvt_kernel<<<dim3((n4 + 255) / 256), dim3(256), 0, stream>>>(e_out, e_bf, n4); }
  gemm_bf16_kernel<2, false, true, false><<<dim3(128 / 32, NNODES / 64), dim3(256), 0, stream>>>(
      e_bf, wa1_t, batt1, z_f, nullptr, NNODES, 128, 256);                         // leaky(e@W_att1+b)

  attdot_kernel<<<dim3(NNODES / 8), dim3(256), 0, stream>>>(z_f, Watt2, batt2, att_f);
  softmax_eg_kernel<<<dim3(1), dim3(1024), 0, stream>>>(att_f, e_out, eg_out);
}
